// dec_i_82257213653481
// MI455X (gfx1250) — matrix operand order repaired and checked by reading, not compiled, not run on hardware
//
#include <hip/hip_runtime.h>
#include <math.h>

#define THREADS 256

// ---------------------------------------------------------------------------
// CDNA5 WMMA types (wave32, v_wmma_f32_16x16x32_bf16)
// ---------------------------------------------------------------------------
typedef __attribute__((ext_vector_type(16))) __bf16       bf16x16;
typedef __attribute__((ext_vector_type(8)))  float        f32x8;
typedef __attribute__((ext_vector_type(4)))  unsigned int u32x4;

union FragBF { u32x4 q[2]; bf16x16 v; };

__device__ __forceinline__ unsigned short f2bf(float f) {
  unsigned int u = __float_as_uint(f);
  u += 0x7FFFu + ((u >> 16) & 1u);           // round-to-nearest-even
  return (unsigned short)(u >> 16);
}

// ---------------------------------------------------------------------------
// im2col (f32 -> bf16), folds nearest-2x upsample and zero padding.
// Output A[M, K] row-major, M = 4*Ho*Wo, K = Cin*ks*ks (multiple of 32 here).
// ---------------------------------------------------------------------------
__global__ void k_im2col(const float* __restrict__ x, unsigned short* __restrict__ A,
                         int Cin, int Hin, int Win, int Ho, int Wo,
                         int ks, int pad, int ups) {
  size_t idx = (size_t)blockIdx.x * THREADS + threadIdx.x;
  int K = Cin * ks * ks;
  size_t total = (size_t)4 * Ho * Wo * (size_t)K;
  if (idx >= total) return;
  size_t m = idx / (size_t)K;
  int k  = (int)(idx - m * (size_t)K);
  int kk = ks * ks;
  int ci = k / kk; int rr = k - ci * kk;
  int kh = rr / ks, kw = rr - kh * ks;
  int HWo = Ho * Wo;
  int b = (int)(m / (size_t)HWo);
  int p = (int)(m - (size_t)b * HWo);
  int oh = p / Wo, ow = p - oh * Wo;
  int Hl = ups ? (Hin << 1) : Hin;
  int Wl = ups ? (Win << 1) : Win;
  int ih = oh + kh - pad, iw = ow + kw - pad;
  float v = 0.0f;
  if (ih >= 0 && ih < Hl && iw >= 0 && iw < Wl) {
    int sh = ups ? (ih >> 1) : ih;
    int sw = ups ? (iw >> 1) : iw;
    v = x[(((size_t)b * Cin + ci) * Hin + sh) * Win + sw];
  }
  A[idx] = f2bf(v);
}

// ---------------------------------------------------------------------------
// Pack weights [Cout, K] (f32, OIHW flattened) into per-lane B-fragment order:
// tile (kt, tn): 512 bf16; lane L, elem e -> K = kt*32 + (L>=16)*16 + e,
//                                           N = tn*16 + (L & 15)
// ---------------------------------------------------------------------------
__global__ void k_pack_w(const float* __restrict__ w, unsigned short* __restrict__ Bp,
                         int N, int K) {
  int tilesN = N >> 4;
  size_t total = (size_t)(K >> 5) * tilesN * 512;
  size_t idx = (size_t)blockIdx.x * THREADS + threadIdx.x;
  if (idx >= total) return;
  int e = (int)(idx & 15);
  int L = (int)((idx >> 4) & 31);
  size_t t = idx >> 9;
  int tn = (int)(t % tilesN);
  int kt = (int)(t / tilesN);
  int n = tn * 16 + (L & 15);
  int k = kt * 32 + ((L >> 4) << 4) + e;
  Bp[idx] = f2bf(w[(size_t)n * K + k]);
}

// ---------------------------------------------------------------------------
// Register-blocked WMMA GEMM: each wave computes a 32x64 macro-tile
// (2 M-tiles x 4 N-tiles). Per 32-deep K step: 2 A-frag + 4 B-frag loads,
// 8 independent v_wmma_f32_16x16x32_bf16. Requires M%32==0, N%64==0
// (true for every conv in this network). Stores NCHW (N = channels).
// ---------------------------------------------------------------------------
__global__ void k_gemm_blk(const unsigned short* __restrict__ A,
                           const unsigned short* __restrict__ Bp,
                           float* __restrict__ out,
                           int M, int N, int K, int HW) {
  int wid  = blockIdx.x * (THREADS >> 5) + (int)(threadIdx.x >> 5);
  int lane = (int)(threadIdx.x & 31);
  int tilesN = N >> 4;
  int tN4 = tilesN >> 2;                 // 64-col macro tiles
  int tM2 = M >> 5;                      // 32-row macro tiles
  int tiles = tM2 * tN4;
  if (wid >= tiles) return;              // wave-uniform: EXEC stays full
  int tm = wid / tN4, tn = wid - tm * tN4;

  f32x8 acc[2][4] = {};
  int row0 = tm * 32 + (lane & 15);
  const unsigned short* arow0 = A + (size_t)row0 * K + ((lane >> 4) << 4);
  const unsigned short* arow1 = arow0 + (size_t)16 * K;
  const unsigned short* bptr  = Bp + ((size_t)tn * 4) * 512 + lane * 16;
  size_t bstep = (size_t)tilesN * 512;

  for (int k0 = 0; k0 < K; k0 += 32) {
    FragBF a0, a1, b0, b1, b2, b3;
    const u32x4* pa0 = (const u32x4*)(arow0 + k0);
    const u32x4* pa1 = (const u32x4*)(arow1 + k0);
    a0.q[0] = pa0[0]; a0.q[1] = pa0[1];   // K 0..15 (upper half-wave +16)
    a1.q[0] = pa1[0]; a1.q[1] = pa1[1];
    const u32x4* pb = (const u32x4*)bptr;
    b0.q[0] = pb[0];  b0.q[1] = pb[1];    // four B loads at 256-elem spacing (tile = 512)
    b1.q[0] = pb[32]; b1.q[1] = pb[33];
    b2.q[0] = pb[64]; b2.q[1] = pb[65];
    b3.q[0] = pb[96]; b3.q[1] = pb[97];
    bptr += bstep;
    acc[0][0] = __builtin_amdgcn_wmma_f32_16x16x32_bf16(false, a0.v, false, b0.v, (short)0, acc[0][0], false, false);
    acc[0][1] = __builtin_amdgcn_wmma_f32_16x16x32_bf16(false, a0.v, false, b1.v, (short)0, acc[0][1], false, false);
    acc[0][2] = __builtin_amdgcn_wmma_f32_16x16x32_bf16(false, a0.v, false, b2.v, (short)0, acc[0][2], false, false);
    acc[0][3] = __builtin_amdgcn_wmma_f32_16x16x32_bf16(false, a0.v, false, b3.v, (short)0, acc[0][3], false, false);
    acc[1][0] = __builtin_amdgcn_wmma_f32_16x16x32_bf16(false, a1.v, false, b0.v, (short)0, acc[1][0], false, false);
    acc[1][1] = __builtin_amdgcn_wmma_f32_16x16x32_bf16(false, a1.v, false, b1.v, (short)0, acc[1][1], false, false);
    acc[1][2] = __builtin_amdgcn_wmma_f32_16x16x32_bf16(false, a1.v, false, b2.v, (short)0, acc[1][2], false, false);
    acc[1][3] = __builtin_amdgcn_wmma_f32_16x16x32_bf16(false, a1.v, false, b3.v, (short)0, acc[1][3], false, false);
  }

  int mbase = tm * 32 + ((lane >> 4) << 3);
  int nlane = lane & 15;
#pragma unroll
  for (int mb = 0; mb < 2; mb++) {
#pragma unroll
    for (int nb = 0; nb < 4; nb++) {
      int n0 = tn * 64 + nb * 16 + nlane;
#pragma unroll
      for (int r = 0; r < 8; r++) {
        int m  = mbase + mb * 16 + r;
        int b_ = m / HW;
        int hw = m - b_ * HW;
        out[((size_t)b_ * N + n0) * HW + hw] = acc[mb][nb][r];
      }
    }
  }
}

// Fallback: one 16x16 tile per wave (kept for shapes without 32/64 blocking).
__global__ void k_gemm16(const unsigned short* __restrict__ A,
                         const unsigned short* __restrict__ Bp,
                         float* __restrict__ out,
                         int M, int N, int K, int HW) {
  int wid  = blockIdx.x * (THREADS >> 5) + (int)(threadIdx.x >> 5);
  int lane = (int)(threadIdx.x & 31);
  int tilesN = N >> 4;
  int tiles  = (M >> 4) * tilesN;
  if (wid >= tiles) return;
  int tm = wid / tilesN, tn = wid - tm * tilesN;

  f32x8 acc = {};
  int rowA = tm * 16 + (lane & 15);
  const unsigned short* arow = A + (size_t)rowA * K + ((lane >> 4) << 4);
  const unsigned short* bptr = Bp + (size_t)tn * 512 + lane * 16;
  size_t bstep = (size_t)tilesN * 512;

  for (int k0 = 0; k0 < K; k0 += 32) {
    FragBF a, b;
    const u32x4* pa = (const u32x4*)(arow + k0);
    a.q[0] = pa[0]; a.q[1] = pa[1];
    const u32x4* pb = (const u32x4*)bptr;
    b.q[0] = pb[0]; b.q[1] = pb[1];
    bptr += bstep;
    acc = __builtin_amdgcn_wmma_f32_16x16x32_bf16(
        false, a.v, false, b.v, (short)0, acc, false, false);
  }

  int n0    = tn * 16 + (lane & 15);
  int mbase = tm * 16 + ((lane >> 4) << 3);
#pragma unroll
  for (int r = 0; r < 8; r++) {
    int m  = mbase + r;
    int b_ = m / HW;
    int hw = m - b_ * HW;
    out[((size_t)b_ * N + n0) * HW + hw] = acc[r];
  }
}

// ---------------------------------------------------------------------------
// Per-(b,c) mean / biased variance over HW
// ---------------------------------------------------------------------------
__global__ void k_stats(const float* __restrict__ x, float* __restrict__ mean,
                        float* __restrict__ var, int HW) {
  int bc = blockIdx.x;
  const float* p = x + (size_t)bc * HW;
  float s = 0.f, s2 = 0.f;
  for (int i = threadIdx.x; i < HW; i += THREADS) { float v = p[i]; s += v; s2 += v * v; }
  __shared__ float sh[2][THREADS];
  sh[0][threadIdx.x] = s; sh[1][threadIdx.x] = s2; __syncthreads();
  for (int st = THREADS >> 1; st > 0; st >>= 1) {
    if ((int)threadIdx.x < st) {
      sh[0][threadIdx.x] += sh[0][threadIdx.x + st];
      sh[1][threadIdx.x] += sh[1][threadIdx.x + st];
    }
    __syncthreads();
  }
  if (threadIdx.x == 0) {
    float m = sh[0][0] / HW;
    mean[bc] = m;
    var[bc]  = fmaxf(sh[1][0] / HW - m * m, 0.f);
  }
}

// InstanceNorm apply (+optional affine) + activation (0 none, 1 relu, 2 leaky)
__global__ void k_in_apply(float* __restrict__ x, const float* __restrict__ mean,
                           const float* __restrict__ var, const float* __restrict__ g,
                           const float* __restrict__ bt, int act,
                           size_t total, int C, int HW) {
  size_t idx = (size_t)blockIdx.x * THREADS + threadIdx.x;
  if (idx >= total) return;
  size_t bc = idx / (size_t)HW;
  int c = (int)(bc % (size_t)C);
  float v = (x[idx] - mean[bc]) * rsqrtf(var[bc] + 1e-5f);
  if (g) v = v * g[c] + bt[c];
  if (act == 1)      v = fmaxf(v, 0.f);
  else if (act == 2) v = (v >= 0.f) ? v : 0.01f * v;
  x[idx] = v;
}

// Style stats over (B=4, HW) per channel: mean, unbiased std + 1e-5
__global__ void k_style(const float* __restrict__ y, float* __restrict__ sm,
                        float* __restrict__ ss, int C, int HW) {
  int c = blockIdx.x;
  float s = 0.f, s2 = 0.f;
  for (int b = 0; b < 4; b++) {
    const float* p = y + ((size_t)b * C + c) * HW;
    for (int i = threadIdx.x; i < HW; i += THREADS) { float v = p[i]; s += v; s2 += v * v; }
  }
  __shared__ float sh[2][THREADS];
  sh[0][threadIdx.x] = s; sh[1][threadIdx.x] = s2; __syncthreads();
  for (int st = THREADS >> 1; st > 0; st >>= 1) {
    if ((int)threadIdx.x < st) {
      sh[0][threadIdx.x] += sh[0][threadIdx.x + st];
      sh[1][threadIdx.x] += sh[1][threadIdx.x + st];
    }
    __syncthreads();
  }
  if (threadIdx.x == 0) {
    float n = (float)(4 * HW);
    float m = sh[0][0] / n;
    float v = fmaxf(sh[1][0] / n - m * m, 0.f);
    sm[c] = m;
    ss[c] = sqrtf(v * n / (n - 1.f)) + 1e-5f;
  }
}

// AdaIN: content unbiased std + 1e-5 (outside sqrt), style stats precomputed
__global__ void k_adain(float* __restrict__ x, const float* __restrict__ mean,
                        const float* __restrict__ var, const float* __restrict__ sm,
                        const float* __restrict__ ss, int C, int HW, size_t total) {
  size_t idx = (size_t)blockIdx.x * THREADS + threadIdx.x;
  if (idx >= total) return;
  size_t bc = idx / (size_t)HW;
  int c = (int)(bc % (size_t)C);
  float cs = sqrtf(var[bc] * ((float)HW / (float)(HW - 1))) + 1e-5f;
  x[idx] = (x[idx] - mean[bc]) / cs * ss[c] + sm[c];
}

// Per-(b,c) avg + max over HW (channel attention pooling)
__global__ void k_pool_bc(const float* __restrict__ x, float* __restrict__ avg,
                          float* __restrict__ mx, int HW) {
  int bc = blockIdx.x;
  const float* p = x + (size_t)bc * HW;
  float s = 0.f, m = -3.4e38f;
  for (int i = threadIdx.x; i < HW; i += THREADS) { float v = p[i]; s += v; m = fmaxf(m, v); }
  __shared__ float shs[THREADS]; __shared__ float shm[THREADS];
  shs[threadIdx.x] = s; shm[threadIdx.x] = m; __syncthreads();
  for (int st = THREADS >> 1; st > 0; st >>= 1) {
    if ((int)threadIdx.x < st) {
      shs[threadIdx.x] += shs[threadIdx.x + st];
      shm[threadIdx.x] = fmaxf(shm[threadIdx.x], shm[threadIdx.x + st]);
    }
    __syncthreads();
  }
  if (threadIdx.x == 0) { avg[bc] = shs[0] / HW; mx[bc] = shm[0]; }
}

// Channel-attention MLP: att = sigmoid(w2 * (relu(w1*avg) + relu(w1*max)))
__global__ void k_chan_mlp(const float* __restrict__ avg, const float* __restrict__ mx,
                           const float* __restrict__ w1, const float* __restrict__ w2,
                           float* __restrict__ att, int C, int Ch) {
  int b = blockIdx.x;
  __shared__ float hsum[64];
  for (int h = threadIdx.x; h < Ch; h += THREADS) {
    float sa = 0.f, sm = 0.f;
    const float* wr = w1 + (size_t)h * C;
    for (int c = 0; c < C; c++) { float wv = wr[c]; sa += wv * avg[b * C + c]; sm += wv * mx[b * C + c]; }
    hsum[h] = fmaxf(sa, 0.f) + fmaxf(sm, 0.f);
  }
  __syncthreads();
  for (int c = threadIdx.x; c < C; c += THREADS) {
    float s = 0.f;
    const float* wr = w2 + (size_t)c * Ch;
    for (int h = 0; h < Ch; h++) s += wr[h] * hsum[h];
    att[b * C + c] = 1.f / (1.f + expf(-s));
  }
}

__global__ void k_scale_ch(float* __restrict__ x, const float* __restrict__ att,
                           int HW, size_t total) {
  size_t idx = (size_t)blockIdx.x * THREADS + threadIdx.x;
  if (idx >= total) return;
  x[idx] *= att[idx / (size_t)HW];
}

// Spatial pooling: [B,2,H,W] = [mean_c, max_c]
__global__ void k_spat_pool(const float* __restrict__ x, float* __restrict__ pool,
                            int C, int HW) {
  int idx = blockIdx.x * THREADS + threadIdx.x;
  if (idx >= 4 * HW) return;
  int b = idx / HW, p = idx - b * HW;
  const float* base = x + (size_t)b * C * HW + p;
  float s = 0.f, m = -3.4e38f;
  for (int c = 0; c < C; c++) { float v = base[(size_t)c * HW]; s += v; m = fmaxf(m, v); }
  pool[((size_t)b * 2 + 0) * HW + p] = s / C;
  pool[((size_t)b * 2 + 1) * HW + p] = m;
}

// 7x7 conv (2->1 ch, pad 3) + sigmoid
__global__ void k_spat_conv(const float* __restrict__ pool, const float* __restrict__ saw,
                            float* __restrict__ satt, int H) {
  int HW = H * H;
  int idx = blockIdx.x * THREADS + threadIdx.x;
  if (idx >= 4 * HW) return;
  int b = idx / HW, p = idx - b * HW;
  int oh = p / H, ow = p - oh * H;
  float acc = 0.f;
  for (int ch = 0; ch < 2; ch++) {
    const float* pm = pool + ((size_t)b * 2 + ch) * HW;
    for (int kh = 0; kh < 7; kh++) {
      int ih = oh + kh - 3; if (ih < 0 || ih >= H) continue;
      for (int kw = 0; kw < 7; kw++) {
        int iw = ow + kw - 3; if (iw < 0 || iw >= H) continue;
        acc += pm[ih * H + iw] * saw[(ch * 7 + kh) * 7 + kw];
      }
    }
  }
  satt[idx] = 1.f / (1.f + expf(-acc));
}

// fused = s * satt; fusion += fused/4; loss += mean|fused - real| / 4
__global__ void k_finalize_mf(const float* __restrict__ s, const float* __restrict__ satt,
                              const float* __restrict__ realx, float* __restrict__ fus,
                              float* __restrict__ loss, int C, int HW, float lscale) {
  size_t total = (size_t)4 * C * HW;
  size_t idx = (size_t)blockIdx.x * THREADS + threadIdx.x;
  float d = 0.f;
  if (idx < total) {
    int b = (int)(idx / ((size_t)C * HW));
    int p = (int)(idx % (size_t)HW);
    float f = s[idx] * satt[(size_t)b * HW + p];
    fus[idx] += 0.25f * f;
    d = fabsf(f - realx[idx]);
  }
  __shared__ float sh[THREADS];
  sh[threadIdx.x] = d; __syncthreads();
  for (int st = THREADS >> 1; st > 0; st >>= 1) {
    if ((int)threadIdx.x < st) sh[threadIdx.x] += sh[threadIdx.x + st];
    __syncthreads();
  }
  if (threadIdx.x == 0) atomicAdd(loss, sh[0] * lscale);
}

// fusion for levels 0,1: average of 4 modality slices (stride S)
__global__ void k_avg4(const float* __restrict__ f, float* __restrict__ fus, size_t S) {
  size_t idx = (size_t)blockIdx.x * THREADS + threadIdx.x;
  if (idx >= S) return;
  fus[idx] = 0.25f * (f[idx] + f[idx + S] + f[idx + 2 * S] + f[idx + 3 * S]);
}

__global__ void k_copy_ch(const float* __restrict__ src, float* __restrict__ dst,
                          int Cs, int Cd, int co, int HW) {
  size_t total = (size_t)4 * Cs * HW;
  size_t idx = (size_t)blockIdx.x * THREADS + threadIdx.x;
  if (idx >= total) return;
  int b = (int)(idx / ((size_t)Cs * HW));
  size_t r = idx - (size_t)b * Cs * HW;
  int c = (int)(r / HW);
  int p = (int)(r - (size_t)c * HW);
  dst[((size_t)b * Cd + co + c) * HW + p] = src[idx];
}

__global__ void k_zero(float* __restrict__ p, size_t n) {
  size_t idx = (size_t)blockIdx.x * THREADS + threadIdx.x;
  if (idx < n) p[idx] = 0.f;
}

// Final 3x3 conv 64->1 + tanh, H=W=128
__global__ void k_conv_out(const float* __restrict__ x, const float* __restrict__ w,
                           float* __restrict__ out) {
  int idx = blockIdx.x * THREADS + threadIdx.x;
  if (idx >= 4 * 16384) return;
  int b = idx >> 14, p = idx & 16383;
  int oh = p >> 7, ow = p & 127;
  float acc = 0.f;
  for (int ci = 0; ci < 64; ci++) {
    const float* xp = x + (((size_t)b * 64 + ci) << 14);
    for (int kh = 0; kh < 3; kh++) {
      int ih = oh + kh - 1; if (ih < 0 || ih > 127) continue;
      for (int kw = 0; kw < 3; kw++) {
        int iw = ow + kw - 1; if (iw < 0 || iw > 127) continue;
        acc += xp[(ih << 7) + iw] * w[(ci * 3 + kh) * 3 + kw];
      }
    }
  }
  out[idx] = tanhf(acc);
}

// ---------------------------------------------------------------------------
// Retrieval: sims[j,b,n] = <q_jb, bank_jn> / max(||bank_jn||, 1e-12)
// One block per (j,n); 65536-long rows.
// ---------------------------------------------------------------------------
__global__ void k_sims(const float* __restrict__ f4, const float* __restrict__ banks,
                       float* __restrict__ sims) {
  int jn = blockIdx.x;
  int j = jn >> 9, n = jn & 511;
  const float* row = banks + ((size_t)j * 512 + n) * 65536;
  const float* q   = f4 + (size_t)j * 4 * 65536;
  float d0 = 0.f, d1 = 0.f, d2 = 0.f, d3 = 0.f, nr = 0.f;
  for (int i = threadIdx.x; i < 65536; i += THREADS) {
    float r = row[i];
    nr += r * r;
    d0 += r * q[i];
    d1 += r * q[65536 + i];
    d2 += r * q[131072 + i];
    d3 += r * q[196608 + i];
  }
  __shared__ float sh[5][THREADS];
  sh[0][threadIdx.x] = d0; sh[1][threadIdx.x] = d1; sh[2][threadIdx.x] = d2;
  sh[3][threadIdx.x] = d3; sh[4][threadIdx.x] = nr;
  __syncthreads();
  for (int st = THREADS >> 1; st > 0; st >>= 1) {
    if ((int)threadIdx.x < st)
      for (int a = 0; a < 5; a++) sh[a][threadIdx.x] += sh[a][threadIdx.x + st];
    __syncthreads();
  }
  if (threadIdx.x == 0) {
    float inv = 1.f / fmaxf(sqrtf(sh[4][0]), 1e-12f);
    for (int b = 0; b < 4; b++) sims[(((size_t)j * 4 + b) << 9) + n] = sh[b][0] * inv;
  }
}

__global__ void k_argmax(const float* __restrict__ sims, int* __restrict__ top) {
  int jb = blockIdx.x;
  const float* s = sims + (size_t)jb * 512;
  float bv = -3.4e38f; int bi = 0x7fffffff;
  for (int i = threadIdx.x; i < 512; i += THREADS) {
    float v = s[i];
    if (v > bv || (v == bv && i < bi)) { bv = v; bi = i; }
  }
  __shared__ float sv[THREADS]; __shared__ int si[THREADS];
  sv[threadIdx.x] = bv; si[threadIdx.x] = bi; __syncthreads();
  for (int st = THREADS >> 1; st > 0; st >>= 1) {
    if ((int)threadIdx.x < st) {
      float ov = sv[threadIdx.x + st]; int oi = si[threadIdx.x + st];
      if (ov > sv[threadIdx.x] || (ov == sv[threadIdx.x] && oi < si[threadIdx.x])) {
        sv[threadIdx.x] = ov; si[threadIdx.x] = oi;
      }
    }
    __syncthreads();
  }
  if (threadIdx.x == 0) top[jb] = si[0];
}

// base[j][b] = banks[IDX=0][top[j][b]]
__global__ void k_gather(const float* __restrict__ banks, const int* __restrict__ top,
                         float* __restrict__ lb0) {
  size_t idx = (size_t)blockIdx.x * THREADS + threadIdx.x;
  if (idx >= (size_t)4 * 4 * 65536) return;
  int j = (int)(idx >> 18);
  size_t r = idx & 262143;
  int b = (int)(r >> 16);
  lb0[idx] = banks[(size_t)top[j * 4 + b] * 65536 + (r & 65535)];
}

// ---------------------------------------------------------------------------
// Host orchestration
// ---------------------------------------------------------------------------
extern "C" void kernel_launch(void* const* d_in, const int* in_sizes, int n_in,
                              void* d_out, int out_size, void* d_ws, size_t ws_size,
                              hipStream_t stream) {
  (void)in_sizes; (void)n_in; (void)out_size; (void)ws_size;
  const int B = 4;

  // ---- input pointers (top level: insertion order; params: jax sorted-key) ----
  const float* F[5];
  for (int i = 0; i < 5; i++) F[i] = (const float*)d_in[i];
  const float* banks = (const float*)d_in[5];

  const float *bup_w[4], *bup_g[4], *bup_bt[4];
  for (int k = 0; k < 4; k++) {           // bup: b, bt, g, w
    int b6 = 6 + 4 * k;
    bup_bt[k] = (const float*)d_in[b6 + 1];
    bup_g[k]  = (const float*)d_in[b6 + 2];
    bup_w[k]  = (const float*)d_in[b6 + 3];
  }
  const float *cv_b1[4], *cv_b2[4], *cv_g1[4], *cv_g2[4], *cv_w1[4], *cv_w2[4];
  for (int k = 0; k < 4; k++) {           // conv: b1, b2, g1, g2, w1, w2
    int b6 = 22 + 6 * k;
    cv_b1[k] = (const float*)d_in[b6 + 0];
    cv_b2[k] = (const float*)d_in[b6 + 1];
    cv_g1[k] = (const float*)d_in[b6 + 2];
    cv_g2[k] = (const float*)d_in[b6 + 3];
    cv_w1[k] = (const float*)d_in[b6 + 4];
    cv_w2[k] = (const float*)d_in[b6 + 5];
  }
  const float *mf_w[3][4], *mf_c1[3][4], *mf_c2[3][4], *mf_sa[3][4];
  for (int ii = 0; ii < 3; ii++)          // mf[i]: b, caw1, caw2, saw, w
    for (int j = 0; j < 4; j++) {
      int b6 = 46 + (ii * 4 + j) * 5;
      mf_c1[ii][j] = (const float*)d_in[b6 + 1];
      mf_c2[ii][j] = (const float*)d_in[b6 + 2];
      mf_sa[ii][j] = (const float*)d_in[b6 + 3];
      mf_w[ii][j]  = (const float*)d_in[b6 + 4];
    }
  const float* out_w = (const float*)d_in[106];
  const float *up_w[4], *up_g[4], *up_bt[4];
  for (int k = 0; k < 4; k++) {           // up: b, bt, g, w
    int b6 = 107 + 4 * k;
    up_bt[k] = (const float*)d_in[b6 + 1];
    up_g[k]  = (const float*)d_in[b6 + 2];
    up_w[k]  = (const float*)d_in[b6 + 3];
  }

  // ---- workspace bump allocator ----
  char* wsb = (char*)d_ws;
  size_t off = 0;
  auto allocf = [&](size_t n) -> float* {
    float* p = (float*)(wsb + off);
    off = (off + n * sizeof(float) + 255) & ~(size_t)255;
    return p;
  };
  auto alloch = [&](size_t n) -> unsigned short* {
    unsigned short* p = (unsigned short*)(wsb + off);
    off = (off + n * 2 + 255) & ~(size_t)255;
    return p;
  };

  float* sims  = allocf(4 * 4 * 512);
  int*   top   = (int*)allocf(16);
  float* lb0   = allocf((size_t)4 * B * 1024 * 64);    // per-j stride 262144
  float* lb1   = allocf((size_t)4 * B * 512 * 256);    // per-j stride 524288
  float* lb2   = allocf((size_t)4 * B * 256 * 1024);   // per-j stride 1048576
  static const int    Cf[5] = {64, 128, 256, 512, 1024};
  static const int    Hf[5] = {128, 64, 32, 16, 8};
  float* fu[5];
  for (int i = 0; i < 5; i++) fu[i] = allocf((size_t)B * Cf[i] * Hf[i] * Hf[i]);
  float* meanb = allocf(4096);
  float* varb  = allocf(4096);
  float* smean = allocf(1024);
  float* sstd  = allocf(1024);
  float* avgp  = allocf(4096);
  float* mxp   = allocf(4096);
  float* attb  = allocf(4096);
  float* pool2 = allocf(8192);
  float* satt  = allocf(4096);
  float* sbuf  = allocf((size_t)B * 256 * 1024);       // max mf feature (i=2)
  float* ubuf  = allocf((size_t)B * 64 * 16384);
  float* tbuf  = allocf((size_t)B * 64 * 16384);
  float* catb  = allocf((size_t)B * 128 * 16384);
  float* xa    = allocf((size_t)B * 64 * 16384);
  float* xb    = allocf((size_t)B * 64 * 16384);
  unsigned short* imA   = alloch((size_t)65536 * 1152);  // max M*K
  unsigned short* packW = alloch((size_t)512 * 9216);    // max Cout*K

  float* outp   = (float*)d_out;
  float* lossp  = outp + 65536;

  auto gridFor = [](size_t n) { return dim3((unsigned)((n + THREADS - 1) / THREADS)); };

  // generic conv (+IN +act), WMMA path
  auto conv_in_act = [&](const float* x, int Cin, int Hin, bool ups, int ks, int pad,
                         const float* w, int Cout, const float* g, const float* bt,
                         int act, float* y) {
    int Ho = ups ? 2 * Hin : Hin;
    int HW = Ho * Ho;
    int K  = Cin * ks * ks;
    size_t M   = (size_t)B * HW;
    size_t tot = M * (size_t)K;
    k_im2col<<<gridFor(tot), THREADS, 0, stream>>>(x, imA, Cin, Hin, Hin, Ho, Ho, ks, pad, ups ? 1 : 0);
    size_t ptot = (size_t)(K >> 5) * (Cout >> 4) * 512;
    k_pack_w<<<gridFor(ptot), THREADS, 0, stream>>>(w, packW, Cout, K);
    if ((M & 31) == 0 && (Cout & 63) == 0) {
      int tiles = (int)(M >> 5) * (Cout >> 6);
      k_gemm_blk<<<dim3((tiles + 7) / 8), THREADS, 0, stream>>>(imA, packW, y, (int)M, Cout, K, HW);
    } else {
      int tiles = (int)(M >> 4) * (Cout >> 4);
      k_gemm16<<<dim3((tiles + 7) / 8), THREADS, 0, stream>>>(imA, packW, y, (int)M, Cout, K, HW);
    }
    k_stats<<<dim3(B * Cout), THREADS, 0, stream>>>(y, meanb, varb, HW);
    size_t n = (size_t)B * Cout * HW;
    k_in_apply<<<gridFor(n), THREADS, 0, stream>>>(y, meanb, varb, g, bt, act, n, Cout, HW);
  };

  // ---- phase 0: retrieval + gather ----
  k_sims<<<dim3(4 * 512), THREADS, 0, stream>>>(F[4], banks, sims);
  k_argmax<<<dim3(16), THREADS, 0, stream>>>(sims, top);
  k_gather<<<gridFor((size_t)4 * 4 * 65536), THREADS, 0, stream>>>(banks, top, lb0);

  // ---- phase 1: bank-upsample chains (only levels used downstream) ----
  for (int j = 0; j < 4; j++) {
    conv_in_act(lb0 + (size_t)j * 262144, 1024, 8, true, 3, 1,
                bup_w[0], 512, bup_g[0], bup_bt[0], 2, lb1 + (size_t)j * 524288);
    conv_in_act(lb1 + (size_t)j * 524288, 512, 16, true, 3, 1,
                bup_w[1], 256, bup_g[1], bup_bt[1], 2, lb2 + (size_t)j * 1048576);
  }

  // ---- zero accumulators ----
  k_zero<<<dim3(1), 1, 0, stream>>>(lossp, 1);
  for (int i = 2; i < 5; i++) {
    size_t n = (size_t)B * Cf[i] * Hf[i] * Hf[i];
    k_zero<<<gridFor(n), THREADS, 0, stream>>>(fu[i], n);
  }

  // ---- phase 2: modality fusion (levels 2,3,4) ----
  for (int ii = 0; ii < 3; ii++) {
    int i = ii + 2;
    int C = Cf[i], H = Hf[i], HW = H * H, Ch = C / 16;
    size_t S = (size_t)B * C * HW;
    const float* ystyle_base = (ii == 0) ? lb2 : (ii == 1) ? lb1 : lb0;
    size_t ystride = (ii == 0) ? 1048576 : (ii == 1) ? 524288 : 262144;
    float lscale = 1.f / ((float)S * 4.f);
    for (int j = 0; j < 4; j++) {
      const float* xin = F[i] + (size_t)j * S;
      // 1x1 conv + IN(no affine) + ReLU  (conv bias cancels under IN)
      conv_in_act(xin, C, H, false, 1, 0, mf_w[ii][j], C, nullptr, nullptr, 1, sbuf);
      // AdaIN
      const float* ys = ystyle_base + (size_t)j * ystride;
      k_style<<<dim3(C), THREADS, 0, stream>>>(ys, smean, sstd, C, HW);
      k_stats<<<dim3(B * C), THREADS, 0, stream>>>(sbuf, meanb, varb, HW);
      k_adain<<<gridFor(S), THREADS, 0, stream>>>(sbuf, meanb, varb, smean, sstd, C, HW, S);
      // channel attention
      k_pool_bc<<<dim3(B * C), THREADS, 0, stream>>>(sbuf, avgp, mxp, HW);
      k_chan_mlp<<<dim3(B), THREADS, 0, stream>>>(avgp, mxp, mf_c1[ii][j], mf_c2[ii][j], attb, C, Ch);
      k_scale_ch<<<gridFor(S), THREADS, 0, stream>>>(sbuf, attb, HW, S);
      // spatial attention + fuse + loss
      k_spat_pool<<<gridFor((size_t)4 * HW), THREADS, 0, stream>>>(sbuf, pool2, C, HW);
      k_spat_conv<<<gridFor((size_t)4 * HW), THREADS, 0, stream>>>(pool2, mf_sa[ii][j], satt, H);
      k_finalize_mf<<<gridFor(S), THREADS, 0, stream>>>(sbuf, satt, F[i] /* modality 0 */,
                                                        fu[i], lossp, C, HW, lscale);
    }
  }
  // fusion levels 0,1: plain average of modalities
  for (int i = 0; i < 2; i++) {
    size_t S = (size_t)B * Cf[i] * Hf[i] * Hf[i];
    k_avg4<<<gridFor(S), THREADS, 0, stream>>>(F[i], fu[i], S);
  }

  // ---- phase 3: decoder ----
  float* xcur = fu[4];
  int Cx = 1024, Hx = 8;
  for (int k = 0; k < 4; k++) {
    int Co = Cx / 2, Ho = 2 * Hx, HW = Ho * Ho;
    conv_in_act(xcur, Cx, Hx, true, 3, 1, up_w[k], Co, up_g[k], up_bt[k], 2, ubuf);
    size_t nc = (size_t)B * Co * HW;
    k_copy_ch<<<gridFor(nc), THREADS, 0, stream>>>(ubuf, catb, Co, 2 * Co, 0, HW);
    k_copy_ch<<<gridFor(nc), THREADS, 0, stream>>>(fu[3 - k], catb, Co, 2 * Co, Co, HW);
    conv_in_act(catb, 2 * Co, Ho, false, 3, 1, cv_w1[k], Co, cv_g1[k], cv_b1[k], 1, tbuf);
    float* xnext = (k & 1) ? xb : xa;
    conv_in_act(tbuf, Co, Ho, false, 3, 1, cv_w2[k], Co, cv_g2[k], cv_b2[k], 2, xnext);
    xcur = xnext; Cx = Co; Hx = Ho;
  }
  k_conv_out<<<gridFor((size_t)4 * 16384), THREADS, 0, stream>>>(xcur, out_w, outp);
}
